// MoEPointwiseConv2d_38543036514531
// MI455X (gfx1250) — compile-verified, matrix operand order checked by reading, not run on hardware
//
#include <hip/hip_runtime.h>
#include <hip/hip_bf16.h>

// ---------------------------------------------------------------------------
// MoE pointwise conv:  out[b] = (sum_e r[b,e] * W[e]) @ x[b]
//   B=64, C_IN=C_OUT=384, E=8, HW=784
// Kernel 1: agg_bf16[b][o][i] = sum_e r[b,e]*W[e][o][i]   (VALU, writes d_ws)
// Kernel 2: batched GEMM via v_wmma_f32_16x16x32_bf16, double-buffered LDS,
//           software-pipelined staging, DS-count-only workgroup barrier,
//           sched_group_barrier-enforced DS-read/WMMA interleave.
// ---------------------------------------------------------------------------

typedef __attribute__((ext_vector_type(16))) __bf16 v16bf;
typedef __attribute__((ext_vector_type(8)))  __bf16 v8bf;
typedef __attribute__((ext_vector_type(4)))  __bf16 v4bf;
typedef __attribute__((ext_vector_type(2)))  __bf16 v2bf;
typedef __attribute__((ext_vector_type(8)))  float  v8f;
typedef __attribute__((ext_vector_type(4)))  float  v4f;

#define NB   64
#define CI   384
#define CO   384
#define NE   8
#define HW   784          // 28*28

#define MBLK 128          // rows of agg per workgroup (8 waves x 16)
#define NBLK 112          // columns of x/out per workgroup (7 wmma tiles)
#define KSTEP 32
#define NKT  (CI / KSTEP) // 12 K tiles
#define LDK  40           // padded K stride in LDS (bf16); 80B keeps 16B align

// sched_group_barrier masks
#define SGB_WMMA   0x008  // MFMA/WMMA
#define SGB_DSREAD 0x100  // DS read

#if defined(__has_builtin) && __has_builtin(__builtin_amdgcn_sched_group_barrier)
#define SGB(mask, size) __builtin_amdgcn_sched_group_barrier((mask), (size), 0)
#else
#define SGB(mask, size)
#endif

static __device__ __forceinline__ v16bf cat8(v8bf lo, v8bf hi) {
    return __builtin_shufflevector(lo, hi, 0,1,2,3,4,5,6,7,8,9,10,11,12,13,14,15);
}

// Workgroup barrier that only drains DScnt (LDS ordering), NOT loadcnt —
// keeps in-flight global prefetch loads pending across the barrier.
static __device__ __forceinline__ void wg_barrier_ds() {
    asm volatile(
        "s_wait_dscnt 0x0\n\t"
        "s_barrier_signal -1\n\t"
        "s_barrier_wait -1"
        ::: "memory");
}

// -------------------------- Kernel 1: aggregate weights ---------------------
__global__ __launch_bounds__(256)
void moe_agg_kernel(const float* __restrict__ rw,     // [B][E]
                    const float* __restrict__ w,      // [E][CO*CI]
                    __bf16*      __restrict__ agg)    // [B][CO*CI]  (bf16)
{
    const int b       = blockIdx.y;
    const size_t base = (size_t)(blockIdx.x * 256 + threadIdx.x) * 4;  // 4 f32 each
    float r[NE];
#pragma unroll
    for (int e = 0; e < NE; ++e) r[e] = rw[b * NE + e];

    v4f acc = {0.f, 0.f, 0.f, 0.f};
#pragma unroll
    for (int e = 0; e < NE; ++e) {
        v4f wv = *(const v4f*)(w + (size_t)e * (CO * CI) + base);
        acc += r[e] * wv;
    }
    v4bf o;
    o.x = (__bf16)acc.x; o.y = (__bf16)acc.y;
    o.z = (__bf16)acc.z; o.w = (__bf16)acc.w;
    *(v4bf*)(agg + (size_t)b * (CO * CI) + base) = o;
}

// -------------------------- Kernel 2: batched WMMA GEMM ---------------------
// grid = (B, CO/MBLK, HW/NBLK) = (64, 3, 7); block = 256 threads = 8 waves.
__global__ __launch_bounds__(256)
void moe_gemm_kernel(const float*  __restrict__ x,    // [B][CI][HW] f32
                     const __bf16* __restrict__ agg,  // [B][CO][CI] bf16
                     float*        __restrict__ out)  // [B][CO][HW] f32
{
    __shared__ __align__(16) __bf16 lds[2][NBLK * LDK];   // double-buffered x tile

    const int b   = blockIdx.x;
    const int m0  = blockIdx.y * MBLK;
    const int hw0 = blockIdx.z * NBLK;

    const int tid  = threadIdx.x;
    const int wave = tid >> 5;        // 0..7
    const int lane = tid & 31;
    const int half = lane >> 4;       // 0/1
    const int l16  = lane & 15;

    // this wave's 16-row A strip (bf16, row-major)
    const int arow = m0 + wave * 16 + l16;
    const __bf16* __restrict__ aggRow =
        agg + (size_t)b * (CO * CI) + (size_t)arow * CI + half * 8;
    const float* __restrict__ xb = x + (size_t)b * (CI * HW) + hw0;

    // per-thread staging coords: 7 (k-pair, n) slots covering the 32x112 tile
    int kk[7], nn[7];
#pragma unroll
    for (int it = 0; it < 7; ++it) {
        const int idx = it * 256 + tid;            // 0..1791
        kk[it] = (idx / NBLK) * 2;                 // even k in [0,32)
        nn[it] = idx - (idx / NBLK) * NBLK;        // n in [0,112)
    }

    v8f acc[7];
#pragma unroll
    for (int n = 0; n < 7; ++n) acc[n] = (v8f){0.f,0.f,0.f,0.f,0.f,0.f,0.f,0.f};

    // ---- prologue: stage tile 0 into buf 0, prefetch A for tile 0 ----
    float fa[7], fb[7];
#pragma unroll
    for (int it = 0; it < 7; ++it) {
        const float* p = xb + (size_t)kk[it] * HW + nn[it];
        fa[it] = p[0];
        fb[it] = p[HW];
    }
    v8bf a_lo = *(const v8bf*)(aggRow + 0);
    v8bf a_hi = *(const v8bf*)(aggRow + 16);
#pragma unroll
    for (int it = 0; it < 7; ++it) {
        *(v2bf*)&lds[0][nn[it] * LDK + kk[it]] = (v2bf){(__bf16)fa[it], (__bf16)fb[it]};
    }

#pragma unroll 2
    for (int i = 0; i < NKT; ++i) {
        const int kc = i * KSTEP;
        const v16bf A = cat8(a_lo, a_hi);

        // ---- issue next tile's prefetch (x pairs + A fragment); no waits ----
        if (i < NKT - 1) {
#pragma unroll
            for (int it = 0; it < 7; ++it) {
                const float* p = xb + (size_t)(kc + KSTEP + kk[it]) * HW + nn[it];
                fa[it] = p[0];
                fb[it] = p[HW];
            }
            a_lo = *(const v8bf*)(aggRow + kc + KSTEP);
            a_hi = *(const v8bf*)(aggRow + kc + KSTEP + 16);
        }

        // DS-only barrier: prefetch loads stay in flight across it
        wg_barrier_ds();

        // ---- 3-deep rotating B-fragment pipeline + 7 WMMAs ----
        // sched_group_barrier pins the order BEFORE regalloc: 3 fragments
        // loaded up front, then WMMA / DS-read alternation -> fragments get
        // distinct registers and each WMMA waits on loads issued 2-3 WMMAs ago.
        {
            const __bf16* __restrict__ cur = lds[i & 1];
            const __bf16* bp = cur + l16 * LDK + half * 8;   // fragment n at bp + n*16*LDK
#define LOADB(n_) cat8(*(const v8bf*)(bp + (n_) * 16 * LDK), \
                       *(const v8bf*)(bp + (n_) * 16 * LDK + 16))
            v16bf B0 = LOADB(0);
            v16bf B1 = LOADB(1);
            v16bf B2 = LOADB(2);
            acc[0] = __builtin_amdgcn_wmma_f32_16x16x32_bf16(
                false, A, false, B0, (short)0, acc[0], false, false);
            B0 = LOADB(3);
            acc[1] = __builtin_amdgcn_wmma_f32_16x16x32_bf16(
                false, A, false, B1, (short)0, acc[1], false, false);
            B1 = LOADB(4);
            acc[2] = __builtin_amdgcn_wmma_f32_16x16x32_bf16(
                false, A, false, B2, (short)0, acc[2], false, false);
            B2 = LOADB(5);
            acc[3] = __builtin_amdgcn_wmma_f32_16x16x32_bf16(
                false, A, false, B0, (short)0, acc[3], false, false);
            B0 = LOADB(6);
            acc[4] = __builtin_amdgcn_wmma_f32_16x16x32_bf16(
                false, A, false, B1, (short)0, acc[4], false, false);
            acc[5] = __builtin_amdgcn_wmma_f32_16x16x32_bf16(
                false, A, false, B2, (short)0, acc[5], false, false);
            acc[6] = __builtin_amdgcn_wmma_f32_16x16x32_bf16(
                false, A, false, B0, (short)0, acc[6], false, false);
#undef LOADB
            // desired schedule: 6 DS reads, (WMMA, 2 DS reads) x4, WMMA x3
            SGB(SGB_DSREAD, 6);
            SGB(SGB_WMMA,   1);
            SGB(SGB_DSREAD, 2);
            SGB(SGB_WMMA,   1);
            SGB(SGB_DSREAD, 2);
            SGB(SGB_WMMA,   1);
            SGB(SGB_DSREAD, 2);
            SGB(SGB_WMMA,   1);
            SGB(SGB_DSREAD, 2);
            SGB(SGB_WMMA,   3);
        }

        // ---- convert + store the prefetched tile into the other buffer ----
        // (s_wait_loadcnt lands here, after the WMMAs -> latency hidden)
        if (i < NKT - 1) {
            __bf16* __restrict__ nxt = lds[(i + 1) & 1];
#pragma unroll
            for (int it = 0; it < 7; ++it) {
                *(v2bf*)&nxt[nn[it] * LDK + kk[it]] =
                    (v2bf){(__bf16)fa[it], (__bf16)fb[it]};
            }
        }
    }

    // ---- epilogue: C/D layout (lane -> N=l16; vgpr r -> M = half*8 + r) ----
    float* __restrict__ ob =
        out + (size_t)b * (CO * HW) + (size_t)(m0 + wave * 16 + half * 8) * HW + hw0 + l16;
#pragma unroll
    for (int n = 0; n < 7; ++n) {
#pragma unroll
        for (int r = 0; r < 8; ++r) {
            ob[(size_t)r * HW + n * 16] = acc[n][r];
        }
    }
}

// ---------------------------------------------------------------------------
extern "C" void kernel_launch(void* const* d_in, const int* in_sizes, int n_in,
                              void* d_out, int out_size, void* d_ws, size_t ws_size,
                              hipStream_t stream) {
    const float* x  = (const float*)d_in[0];   // [64,384,28,28]
    const float* rw = (const float*)d_in[1];   // [64,8]
    const float* w  = (const float*)d_in[2];   // [8,384,384]
    float* out      = (float*)d_out;           // [64,384,28,28]
    __bf16* agg     = (__bf16*)d_ws;           // [64,384,384] bf16 = 18.9 MB

    moe_agg_kernel<<<dim3((CO * CI) / (256 * 4), NB), 256, 0, stream>>>(rw, w, agg);
    moe_gemm_kernel<<<dim3(NB, CO / MBLK, HW / NBLK), 256, 0, stream>>>(x, agg, out);
}
